// ReprogrammingLayer_36507222016547
// MI455X (gfx1250) — matrix operand order repaired and checked by reading, not compiled, not run on hardware
//
#include <hip/hip_runtime.h>
#include <hip/hip_bf16.h>
#include <stdint.h>

// MI455X / gfx1250, wave32. bf16 WMMA (16x16x32) with f32 accumulation.
// LDS staging now uses CDNA5 async direct-to-LDS copies (ASYNCcnt) instead of
// global->VGPR->ds_store, plus global_prefetch for streamed weight tiles.
//
// Pipeline:
//   1) convert fp32 inputs -> bf16 workspace copies (prototypes zero-padded to 1024 rows)
//   2) Q = X @ Wq^T + bq            (bf16 out)      [8192 x 4096]
//   3) K = proto @ Wk^T + bk        (bf16 out)      [1024 x 4096]
//   4) V = proto @ Wv^T + bv        (bf16 out)      [1024 x 4096]
//   5) fused flash-style cross-attention over P     (bf16 out)  [8192 x 4096]
//   6) out = O @ Wo^T + bo          (f32 out to d_out)

typedef __bf16 bf16_t;
typedef __attribute__((ext_vector_type(16))) __bf16 v16bf;
typedef __attribute__((ext_vector_type(8)))  __bf16 v8bf;
typedef __attribute__((ext_vector_type(8)))  float   v8f;

union V16 { v16bf v; v8bf h[2]; };

__device__ __forceinline__ v8f v8f_zero() {
  v8f z;
#pragma unroll
  for (int i = 0; i < 8; ++i) z[i] = 0.0f;
  return z;
}

#define WMMA_BF16(A, B, C) \
  __builtin_amdgcn_wmma_f32_16x16x32_bf16(false, (A), false, (B), (short)0, (C), false, false)

// Async direct global->LDS copy, 16B per lane. VDST = LDS byte address (low 32
// bits of the generic pointer), VADDR = 64-bit global address. ASYNCcnt++.
__device__ __forceinline__ void async_ld_b128(void* lds, const void* g) {
  unsigned l = (unsigned)(uintptr_t)lds;
  asm volatile("global_load_async_to_lds_b128 %0, %1, off"
               :: "v"(l), "v"(g) : "memory");
}
__device__ __forceinline__ void wait_async0() {
  asm volatile("s_wait_asynccnt 0x0" ::: "memory");
}

// ---------------------------------------------------------------------------
// fp32 -> bf16 conversion kernels
// ---------------------------------------------------------------------------
__global__ void cvt_f32_bf16(const float* __restrict__ in, bf16_t* __restrict__ out, size_t n) {
  size_t i = (size_t)blockIdx.x * 256 + threadIdx.x;
  if (i < n) out[i] = (bf16_t)in[i];
}

// prototypes: [1000,4096] f32 -> [1024,4096] bf16 with zero padding rows
__global__ void cvt_proto(const float* __restrict__ in, bf16_t* __restrict__ out) {
  size_t i = (size_t)blockIdx.x * 256 + threadIdx.x;   // over 1024*4096
  size_t row = i >> 12;
  out[i] = (row < 1000) ? (bf16_t)in[i] : (bf16_t)0.0f;
}

// ---------------------------------------------------------------------------
// NT GEMM: C[m,n] = sum_k A[m,k] * W[n,k] + bias[n]
//   A: M x K bf16 row-major, W: N x K bf16 row-major.
//   Block tile 128x128x32, 256 threads = 8 waves in a 4(M) x 2(N) grid,
//   each wave computes 32x64 via 2x4 WMMA fragments.
// ---------------------------------------------------------------------------
template <typename OutT>
__global__ __launch_bounds__(256) void gemm_nt_bf16(const bf16_t* __restrict__ A,
                                                    const bf16_t* __restrict__ W,
                                                    const float*  __restrict__ bias,
                                                    OutT* __restrict__ C,
                                                    int M, int Nt, int K) {
  constexpr int LDT = 40;                 // padded LDS stride (bf16 elems), keeps 16B align
  __shared__ bf16_t As[128 * LDT];
  __shared__ bf16_t Bs[128 * LDT];

  const int tid  = threadIdx.x;
  const int wv   = tid >> 5;
  const int lane = tid & 31;
  const int lm   = lane & 15;
  const int lh   = lane >> 4;
  const int mw   = wv & 3;                // wave position in M (0..3)
  const int nw   = wv >> 2;               // wave position in N (0..1)
  const size_t m0 = (size_t)blockIdx.y * 128;
  const size_t n0 = (size_t)blockIdx.x * 128;
  const int lr = tid >> 1;                // 0..127: tile row this thread stages
  const int lc = (tid & 1) * 16;          // 0 or 16: k-half this thread stages

  const bf16_t* arow = A + (m0 + (size_t)lr) * (size_t)K + lc;
  const bf16_t* wrow = W + (n0 + (size_t)lr) * (size_t)K + lc;
  bf16_t* asm0 = &As[lr * LDT + lc];
  bf16_t* bsm0 = &Bs[lr * LDT + lc];

  v8f acc[2][4];
#pragma unroll
  for (int i = 0; i < 2; ++i)
#pragma unroll
    for (int j = 0; j < 4; ++j) acc[i][j] = v8f_zero();

  for (int k0 = 0; k0 < K; k0 += 32) {
    __syncthreads();
    // stage A tile (128x32) and W tile (128x32): async direct-to-LDS, 2x16B each
    async_ld_b128(asm0,     arow + k0);
    async_ld_b128(asm0 + 8, arow + k0 + 8);
    async_ld_b128(bsm0,     wrow + k0);
    async_ld_b128(bsm0 + 8, wrow + k0 + 8);
    // prefetch the tiles two chunks ahead into L2
    if (k0 + 64 < K) {
      __builtin_prefetch(arow + k0 + 64, 0, 1);
      __builtin_prefetch(wrow + k0 + 64, 0, 1);
    }
    wait_async0();
    __syncthreads();

    // A fragment: lanes 0-15 rows M=0..15 K=0..7/16..23, lanes 16-31 K=8..15/24..31
    V16 af[2], bw[4];
#pragma unroll
    for (int i = 0; i < 2; ++i) {
      const int mr = mw * 32 + i * 16 + lm;
      af[i].h[0] = *(const v8bf*)&As[mr * LDT + lh * 8];
      af[i].h[1] = *(const v8bf*)&As[mr * LDT + lh * 8 + 16];
    }
    // B fragment: lanes 0-15 hold K=0..7/16..23 of their column, lanes 16-31 K=8..15/24..31
    // B[k][n] = W[n][k] -> bf16 k=8h..8h+7 and 16+8h..16+8h+7 from the weight row
#pragma unroll
    for (int j = 0; j < 4; ++j) {
      const int nr = nw * 64 + j * 16 + lm;
      bw[j].h[0] = *(const v8bf*)&Bs[nr * LDT + lh * 8];
      bw[j].h[1] = *(const v8bf*)&Bs[nr * LDT + lh * 8 + 16];
    }
#pragma unroll
    for (int i = 0; i < 2; ++i)
#pragma unroll
      for (int j = 0; j < 4; ++j)
        acc[i][j] = WMMA_BF16(af[i].v, bw[j].v, acc[i][j]);
  }

  // epilogue: C layout = VGPR r -> row (r + 8*laneHalf), col = lane&15
#pragma unroll
  for (int i = 0; i < 2; ++i) {
#pragma unroll
    for (int j = 0; j < 4; ++j) {
      const size_t n = n0 + (size_t)(nw * 64 + j * 16 + lm);
      const float bn = bias[n];
#pragma unroll
      for (int r = 0; r < 8; ++r) {
        const size_t m = m0 + (size_t)(mw * 32 + i * 16 + r + 8 * lh);
        C[m * (size_t)Nt + n] = (OutT)(acc[i][j][r] + bn);
      }
    }
  }
}

// ---------------------------------------------------------------------------
// Fused cross-attention (flash-style over P).
//   grid = (N/128, H=32, B=16), 256 threads = 8 waves; each wave owns 16 query
//   rows, streams prototypes in chunks of 32 (padded P=1024, cols>=1000 masked).
// ---------------------------------------------------------------------------
__global__ __launch_bounds__(256) void attn_kernel(const bf16_t* __restrict__ Qb,
                                                   const bf16_t* __restrict__ Kb,
                                                   const bf16_t* __restrict__ Vb,
                                                   bf16_t* __restrict__ Ob) {
  __shared__ bf16_t Ksm[32 * 136];   // K chunk  [32 p][128 d], padded stride
  __shared__ bf16_t Vts[128 * 40];   // V chunk transposed [128 d][32 p]
  __shared__ bf16_t Pl[8 * 512];     // per-wave probability tile 16x32 (A-layout staging)

  const int tid  = threadIdx.x;
  const int wv   = tid >> 5;
  const int lane = tid & 31;
  const int lm   = lane & 15;
  const int lh   = lane >> 4;
  const int h    = blockIdx.y;
  const int b    = blockIdx.z;
  const int n0   = blockIdx.x * 128 + wv * 16;

  // Q fragments for the whole head dim (dh=128 -> 4 K-steps), direct from global
  V16 aq[4];
  const size_t qrow = ((size_t)(b * 512 + n0 + lm)) * 4096 + (size_t)h * 128;
#pragma unroll
  for (int kc = 0; kc < 4; ++kc) {
    aq[kc].h[0] = *(const v8bf*)(Qb + qrow + kc * 32 + lh * 8);
    aq[kc].h[1] = *(const v8bf*)(Qb + qrow + kc * 32 + lh * 8 + 16);
  }

  v8f o[8];
#pragma unroll
  for (int t = 0; t < 8; ++t) o[t] = v8f_zero();
  float mrow[8], lrow[8];
#pragma unroll
  for (int r = 0; r < 8; ++r) { mrow[r] = -3.0e38f; lrow[r] = 0.0f; }

  const int lr  = tid >> 3;          // 0..31 chunk row this thread stages
  const int c16 = (tid & 7) * 16;    // 0..112 d-offset this thread stages
  bf16_t* Pw = &Pl[wv * 512];
  const bf16_t* krow = Kb + (size_t)lr * 4096 + (size_t)h * 128 + c16;
  const bf16_t* vrow = Vb + (size_t)lr * 4096 + (size_t)h * 128 + c16;
  bf16_t* ksm0 = &Ksm[lr * 136 + c16];

  for (int p0 = 0; p0 < 1024; p0 += 32) {
    __syncthreads();
    {
      const size_t g = (size_t)p0 * 4096;
      // K chunk: async direct-to-LDS
      async_ld_b128(ksm0,     krow + g);
      async_ld_b128(ksm0 + 8, krow + g + 8);
      // V chunk: needs a transpose, so route through VGPRs
      v8bf w0 = *(const v8bf*)(vrow + g);
      v8bf w1 = *(const v8bf*)(vrow + g + 8);
#pragma unroll
      for (int e = 0; e < 8; ++e) {
        Vts[(c16 + e) * 40 + lr]     = w0[e];
        Vts[(c16 + 8 + e) * 40 + lr] = w1[e];
      }
      wait_async0();
    }
    __syncthreads();

    // scores S = Q . K^T for two 16x16 p-tiles
    v8f s0 = v8f_zero(), s1 = v8f_zero();
#pragma unroll
    for (int kc = 0; kc < 4; ++kc) {
      V16 b0, b1;
      b0.h[0] = *(const v8bf*)&Ksm[lm * 136 + kc * 32 + lh * 8];
      b0.h[1] = *(const v8bf*)&Ksm[lm * 136 + kc * 32 + lh * 8 + 16];
      b1.h[0] = *(const v8bf*)&Ksm[(lm + 16) * 136 + kc * 32 + lh * 8];
      b1.h[1] = *(const v8bf*)&Ksm[(lm + 16) * 136 + kc * 32 + lh * 8 + 16];
      s0 = WMMA_BF16(aq[kc].v, b0.v, s0);
      s1 = WMMA_BF16(aq[kc].v, b1.v, s1);
    }

    // online softmax over this 32-wide p chunk
    const bool mask0 = (p0 + lm) >= 1000;
    const bool mask1 = (p0 + 16 + lm) >= 1000;
#pragma unroll
    for (int r = 0; r < 8; ++r) {
      float v0 = s0[r] * 0.08838834764831845f;   // 1/sqrt(128)
      float v1 = s1[r] * 0.08838834764831845f;
      if (mask0) v0 = -1.0e30f;
      if (mask1) v1 = -1.0e30f;
      float mx = fmaxf(v0, v1);
      mx = fmaxf(mx, __shfl_xor(mx, 1));
      mx = fmaxf(mx, __shfl_xor(mx, 2));
      mx = fmaxf(mx, __shfl_xor(mx, 4));
      mx = fmaxf(mx, __shfl_xor(mx, 8));
      const float mnew  = fmaxf(mrow[r], mx);
      const float alpha = __expf(mrow[r] - mnew);
      mrow[r] = mnew;
      v0 = __expf(v0 - mnew);
      v1 = __expf(v1 - mnew);
      float rs = v0 + v1;
      rs += __shfl_xor(rs, 1);
      rs += __shfl_xor(rs, 2);
      rs += __shfl_xor(rs, 4);
      rs += __shfl_xor(rs, 8);
      lrow[r] = lrow[r] * alpha + rs;
#pragma unroll
      for (int t = 0; t < 8; ++t) o[t][r] *= alpha;
      // stage probabilities as a 16x32 row-major tile (wave-local)
      Pw[(r + 8 * lh) * 32 + lm]      = (bf16_t)v0;
      Pw[(r + 8 * lh) * 32 + 16 + lm] = (bf16_t)v1;
    }
    asm volatile("s_wait_dscnt 0" ::: "memory");   // wave-local LDS store->load ordering

    // O += P(16x32) . V(32x128)
    V16 ap;
    ap.h[0] = *(const v8bf*)&Pw[lm * 32 + lh * 8];
    ap.h[1] = *(const v8bf*)&Pw[lm * 32 + lh * 8 + 16];
#pragma unroll
    for (int dt = 0; dt < 8; ++dt) {
      V16 bvf;
      bvf.h[0] = *(const v8bf*)&Vts[(dt * 16 + lm) * 40 + lh * 8];
      bvf.h[1] = *(const v8bf*)&Vts[(dt * 16 + lm) * 40 + lh * 8 + 16];
      o[dt] = WMMA_BF16(ap.v, bvf.v, o[dt]);
    }
  }

  // normalize and write back in [b*512+n][h*128+d] layout (bf16 for final GEMM)
#pragma unroll
  for (int r = 0; r < 8; ++r) {
    const float inv = 1.0f / lrow[r];
    const size_t row = (size_t)(b * 512 + n0 + r + 8 * lh);
#pragma unroll
    for (int dt = 0; dt < 8; ++dt) {
      Ob[row * 4096 + (size_t)(h * 128 + dt * 16 + lm)] = (bf16_t)(o[dt][r] * inv);
    }
  }
}

// ---------------------------------------------------------------------------
// launch
// ---------------------------------------------------------------------------
extern "C" void kernel_launch(void* const* d_in, const int* in_sizes, int n_in,
                              void* d_out, int out_size, void* d_ws, size_t ws_size,
                              hipStream_t stream) {
  (void)in_sizes; (void)n_in; (void)out_size; (void)ws_size;

  const float* patch = (const float*)d_in[0];   // [16,512,1024]
  const float* proto = (const float*)d_in[1];   // [1000,4096]
  const float* Wq = (const float*)d_in[2];      // [4096,1024]
  const float* bq = (const float*)d_in[3];
  const float* Wk = (const float*)d_in[4];      // [4096,4096]
  const float* bk = (const float*)d_in[5];
  const float* Wv = (const float*)d_in[6];
  const float* bv = (const float*)d_in[7];
  const float* Wo = (const float*)d_in[8];
  const float* bo = (const float*)d_in[9];
  float* out = (float*)d_out;                   // [16,512,4096]

  char* ws = (char*)d_ws;
  bf16_t* Xb    = (bf16_t*)ws;                       ws += (size_t)8192 * 1024 * 2;
  bf16_t* Protb = (bf16_t*)ws;                       ws += (size_t)1024 * 4096 * 2;
  bf16_t* Wqb   = (bf16_t*)ws;                       ws += (size_t)4096 * 1024 * 2;
  bf16_t* Wkb   = (bf16_t*)ws;                       ws += (size_t)4096 * 4096 * 2;
  bf16_t* Wvb   = (bf16_t*)ws;                       ws += (size_t)4096 * 4096 * 2;
  bf16_t* Wob   = (bf16_t*)ws;                       ws += (size_t)4096 * 4096 * 2;
  bf16_t* Qbf   = (bf16_t*)ws;                       ws += (size_t)8192 * 4096 * 2;
  bf16_t* Kbf   = (bf16_t*)ws;                       ws += (size_t)1024 * 4096 * 2;
  bf16_t* Vbf   = (bf16_t*)ws;                       ws += (size_t)1024 * 4096 * 2;
  bf16_t* Obf   = (bf16_t*)ws;                       ws += (size_t)8192 * 4096 * 2;

  const dim3 blk(256);

  // 1) conversions
  cvt_f32_bf16<<<dim3((unsigned)(((size_t)8192 * 1024) / 256)), blk, 0, stream>>>(patch, Xb, (size_t)8192 * 1024);
  cvt_f32_bf16<<<dim3((unsigned)(((size_t)4096 * 1024) / 256)), blk, 0, stream>>>(Wq, Wqb, (size_t)4096 * 1024);
  cvt_f32_bf16<<<dim3((unsigned)(((size_t)4096 * 4096) / 256)), blk, 0, stream>>>(Wk, Wkb, (size_t)4096 * 4096);
  cvt_f32_bf16<<<dim3((unsigned)(((size_t)4096 * 4096) / 256)), blk, 0, stream>>>(Wv, Wvb, (size_t)4096 * 4096);
  cvt_f32_bf16<<<dim3((unsigned)(((size_t)4096 * 4096) / 256)), blk, 0, stream>>>(Wo, Wob, (size_t)4096 * 4096);
  cvt_proto<<<dim3((unsigned)(((size_t)1024 * 4096) / 256)), blk, 0, stream>>>(proto, Protb);

  // 2) Q = X @ Wq^T + bq   [8192 x 4096], K=1024
  gemm_nt_bf16<bf16_t><<<dim3(32, 64), blk, 0, stream>>>(Xb, Wqb, bq, Qbf, 8192, 4096, 1024);
  // 3) K = proto @ Wk^T + bk   [1024 x 4096], K=4096
  gemm_nt_bf16<bf16_t><<<dim3(32, 8), blk, 0, stream>>>(Protb, Wkb, bk, Kbf, 1024, 4096, 4096);
  // 4) V = proto @ Wv^T + bv
  gemm_nt_bf16<bf16_t><<<dim3(32, 8), blk, 0, stream>>>(Protb, Wvb, bv, Vbf, 1024, 4096, 4096);

  // 5) fused attention
  attn_kernel<<<dim3(4, 32, 16), blk, 0, stream>>>(Qbf, Kbf, Vbf, Obf);

  // 6) out = O @ Wo^T + bo  (f32 output)
  gemm_nt_bf16<float><<<dim3(32, 64), blk, 0, stream>>>(Obf, Wob, bo, out, 8192, 4096, 4096);
}
